// OverdampedIVPShooterLoss_80204219285758
// MI455X (gfx1250) — compile-verified, matrix operand order checked by reading, not run on hardware
//
#include <hip/hip_runtime.h>
#include <hip/hip_bf16.h>

typedef __attribute__((ext_vector_type(16))) __bf16 v16bf;
typedef __attribute__((ext_vector_type(8)))  float  v8f;

#define BM 128
#define BN 128
#define BK 32
#define BKP 40  // padded K stride (elements) -> 80B rows, 16B-aligned, conflict-free b128
#define DT 0.1f

// ---------- helpers ----------
__device__ __forceinline__ unsigned short f2bf_bits(float f) {
    union { float f; unsigned u; } x; x.f = f;
    unsigned r = x.u + 0x7FFFu + ((x.u >> 16) & 1u);   // round-to-nearest-even
    return (unsigned short)(r >> 16);
}

union FragBF { uint4 q[2]; v16bf v; };

// ---------- prep: f32 -> bf16 bulk convert ----------
__global__ __launch_bounds__(256)
void convert_f32_bf16(const float* __restrict__ in, unsigned short* __restrict__ out, int n)
{
    int i = (blockIdx.x * 256 + threadIdx.x) * 4;
    if (i < n) {
        float4 v = *(const float4*)(in + i);
        ushort4 p;
        p.x = f2bf_bits(v.x); p.y = f2bf_bits(v.y);
        p.z = f2bf_bits(v.z); p.w = f2bf_bits(v.w);
        *(ushort4*)(out + i) = p;
    }
}

// ---------- prep: W f32 [K,N] -> Wt bf16 [N,K] (LDS-tiled transpose) ----------
__global__ __launch_bounds__(256)
void transpose_f32_bf16(const float* __restrict__ in, unsigned short* __restrict__ out,
                        int K, int N)
{
    __shared__ float t[32][33];
    int k0 = blockIdx.x * 32;
    int n0 = blockIdx.y * 32;
    int tr = threadIdx.x >> 5;   // 0..7
    int tc = threadIdx.x & 31;
    #pragma unroll
    for (int i = 0; i < 4; ++i)
        t[tr + i * 8][tc] = in[(size_t)(k0 + tr + i * 8) * N + n0 + tc];
    __syncthreads();
    #pragma unroll
    for (int i = 0; i < 4; ++i)
        out[(size_t)(n0 + tr + i * 8) * K + k0 + tc] = f2bf_bits(t[tc][tr + i * 8]);
}

// ---------- unified GEMM: A bf16 [M,K] x Wt bf16 [N,K] + bias ----------
// Async-to-LDS double-buffered pipeline, one workgroup barrier per K-tile.
// mode 0: OutBf = bf16(tanh(v))
// mode 1: OutF = v ; if OutBf != null also OutBf = bf16(v)
// mode 2: zn = OutF + DT*v ; OutF = zn ; OutBf = bf16(zn)     (overdamped Euler)
__global__ __launch_bounds__(256)
void gemm_bf16(const unsigned short* __restrict__ A, const unsigned short* __restrict__ Wt,
               const float* __restrict__ bias, unsigned short* __restrict__ OutBf,
               float* __restrict__ OutF, int M, int K, int N, int mode)
{
    __shared__ unsigned short lA[2][BM * BKP];
    __shared__ unsigned short lB[2][BN * BKP];

    const int bm   = blockIdx.x * BM;
    const int bn   = blockIdx.y * BN;
    const int tid  = threadIdx.x;
    const int lane = tid & 31;
    const int wave = tid >> 5;
    const int wr   = wave >> 2;   // 0..1 -> 64-row slab
    const int wc   = wave & 3;    // 0..3 -> 32-col slab
    const int m15  = lane & 15;
    const int klo  = ((lane >> 4) & 1) * 8;   // K element offset: 0 or 8
    const int nk   = K / BK;

    // per-thread tile-copy slots: 2 x 16B per matrix per tile
    const unsigned short* pA[2];
    const unsigned short* pB[2];
    unsigned lo[2];
    #pragma unroll
    for (int i = 0; i < 2; ++i) {
        int idx = tid + i * 256;           // 0..511 slots of 8 bf16
        int r   = idx >> 2;                // 0..127
        int g   = idx & 3;                 // group of 8 elements
        pA[i]  = A  + (size_t)(bm + r) * K + g * 8;
        pB[i]  = Wt + (size_t)(bn + r) * K + g * 8;
        lo[i]  = r * BKP + g * 8;
    }

    auto issue_tile = [&](int t, int buf) {
        #pragma unroll
        for (int i = 0; i < 2; ++i) {
            asm volatile("global_load_async_to_lds_b128 %0, %1, off"
                         :: "v"((unsigned)(size_t)&lA[buf][lo[i]]), "v"(pA[i]) : "memory");
            asm volatile("global_load_async_to_lds_b128 %0, %1, off"
                         :: "v"((unsigned)(size_t)&lB[buf][lo[i]]), "v"(pB[i]) : "memory");
            if (t + 2 < nk) {   // L2 prefetch one stage beyond the double buffer
                __builtin_prefetch(pA[i] + 2 * BK, 0, 1);
                __builtin_prefetch(pB[i] + 2 * BK, 0, 1);
            }
            pA[i] += BK;
            pB[i] += BK;
        }
    };

    v8f acc[4][2];
    #pragma unroll
    for (int i = 0; i < 4; ++i)
        #pragma unroll
        for (int j = 0; j < 2; ++j)
            #pragma unroll
            for (int e = 0; e < 8; ++e) acc[i][j][e] = 0.f;

    issue_tile(0, 0);

    for (int t = 0; t < nk; ++t) {
        const int cur = t & 1;
        asm volatile("s_wait_asynccnt 0x0" ::: "memory");
        __syncthreads();
        if (t + 1 < nk) issue_tile(t + 1, cur ^ 1);

        const unsigned short* curA = &lA[cur][0];
        const unsigned short* curB = &lB[cur][0];
        FragBF a[4], b[2];
        #pragma unroll
        for (int mt = 0; mt < 4; ++mt) {
            int row = wr * 64 + mt * 16 + m15;
            a[mt].q[0] = *(const uint4*)&curA[row * BKP + klo];
            a[mt].q[1] = *(const uint4*)&curA[row * BKP + klo + 16];
        }
        #pragma unroll
        for (int nt = 0; nt < 2; ++nt) {
            int col = wc * 32 + nt * 16 + m15;
            b[nt].q[0] = *(const uint4*)&curB[col * BKP + klo];
            b[nt].q[1] = *(const uint4*)&curB[col * BKP + klo + 16];
        }
        #pragma unroll
        for (int mt = 0; mt < 4; ++mt)
            #pragma unroll
            for (int nt = 0; nt < 2; ++nt)
                acc[mt][nt] = __builtin_amdgcn_wmma_f32_16x16x32_bf16(
                    false, a[mt].v, false, b[nt].v, (short)0, acc[mt][nt], false, false);
    }

    #pragma unroll
    for (int mt = 0; mt < 4; ++mt) {
        #pragma unroll
        for (int nt = 0; nt < 2; ++nt) {
            int row0 = bm + wr * 64 + mt * 16 + ((lane >> 4) & 1) * 8;
            int col  = bn + wc * 32 + nt * 16 + m15;
            float bv = bias[col];
            #pragma unroll
            for (int r = 0; r < 8; ++r) {
                float v = acc[mt][nt][r] + bv;
                size_t idx = (size_t)(row0 + r) * N + col;
                if (mode == 0) {
                    OutBf[idx] = f2bf_bits(tanhf(v));
                } else if (mode == 1) {
                    OutF[idx] = v;
                    if (OutBf) OutBf[idx] = f2bf_bits(v);
                } else {
                    float zn = OutF[idx] + DT * v;
                    OutF[idx]  = zn;
                    OutBf[idx] = f2bf_bits(zn);
                }
            }
        }
    }
}

// ---------- per-row squared distance to z_e, running min (one wave32 per row) ----------
__global__ __launch_bounds__(256)
void dist_min_kernel(const float* __restrict__ z, const float* __restrict__ ze,
                     float* __restrict__ mind)
{
    int row  = blockIdx.x * 8 + (threadIdx.x >> 5);
    int lane = threadIdx.x & 31;
    const float* zr = z  + (size_t)row * 256;
    const float* er = ze + (size_t)row * 256;
    float s = 0.f;
    #pragma unroll
    for (int i = 0; i < 8; ++i) {
        float d = zr[lane + i * 32] - er[lane + i * 32];
        s += d * d;
    }
    #pragma unroll
    for (int off = 16; off > 0; off >>= 1) s += __shfl_xor(s, off, 32);
    if (lane == 0) mind[row] = fminf(mind[row], s);
}

__global__ void init_min_kernel(float* mind, int n) {
    int i = blockIdx.x * blockDim.x + threadIdx.x;
    if (i < n) mind[i] = 3.402823466e+38f;
}

__global__ __launch_bounds__(256)
void mean_kernel(const float* __restrict__ mind, float* __restrict__ out, int n)
{
    __shared__ float sm[256];
    float s = 0.f;
    for (int i = threadIdx.x; i < n; i += 256) s += mind[i];
    sm[threadIdx.x] = s;
    __syncthreads();
    for (int st = 128; st > 0; st >>= 1) {
        if (threadIdx.x < st) sm[threadIdx.x] += sm[threadIdx.x + st];
        __syncthreads();
    }
    if (threadIdx.x == 0) out[0] = sm[0] / (float)n;   // WEIGHT == 1.0
}

// ---------- launch ----------
extern "C" void kernel_launch(void* const* d_in, const int* in_sizes, int n_in,
                              void* d_out, int out_size, void* d_ws, size_t ws_size,
                              hipStream_t stream) {
    const int B = 4096, Dx = 1024, H = 1024, D = 256, STEPS = 15;

    const float* x_start = (const float*)d_in[0];
    const float* x_end   = (const float*)d_in[1];
    const float* enc_w1  = (const float*)d_in[2];
    const float* enc_b1  = (const float*)d_in[3];
    const float* enc_w2  = (const float*)d_in[4];
    const float* enc_b2  = (const float*)d_in[5];
    const float* wind_w1 = (const float*)d_in[6];
    const float* wind_b1 = (const float*)d_in[7];
    const float* wind_w2 = (const float*)d_in[8];
    const float* wind_b2 = (const float*)d_in[9];

    // workspace carve-up (bf16 sections first, then f32)
    unsigned short* xbf  = (unsigned short*)d_ws;       // [B,Dx]  8 MB (reused for both x's)
    unsigned short* h    = xbf  + (size_t)B * Dx;       // [B,H]   8 MB
    unsigned short* zbf  = h    + (size_t)B * H;        // [B,D]   2 MB
    unsigned short* ew1t = zbf  + (size_t)B * D;        // [H,Dx]  2 MB
    unsigned short* ew2t = ew1t + (size_t)H * Dx;       // [D,H]   0.5 MB
    unsigned short* ww1t = ew2t + (size_t)D * H;        // [H,D]   0.5 MB
    unsigned short* ww2t = ww1t + (size_t)H * D;        // [D,H]   0.5 MB
    float* z    = (float*)(ww2t + (size_t)D * H);       // [B,D]   4 MB
    float* ze   = z  + (size_t)B * D;                   // [B,D]   4 MB
    float* mind = ze + (size_t)B * D;                   // [B]

    dim3 blk(256);
    dim3 g1(B / BM, H / BN);   // N = 1024
    dim3 g2(B / BM, D / BN);   // N = 256

    // one-time prep: transpose+convert weights, convert activations
    transpose_f32_bf16<<<dim3(Dx/32, H/32), blk, 0, stream>>>(enc_w1,  ew1t, Dx, H);
    transpose_f32_bf16<<<dim3(H/32,  D/32), blk, 0, stream>>>(enc_w2,  ew2t, H,  D);
    transpose_f32_bf16<<<dim3(D/32,  H/32), blk, 0, stream>>>(wind_w1, ww1t, D,  H);
    transpose_f32_bf16<<<dim3(H/32,  D/32), blk, 0, stream>>>(wind_w2, ww2t, H,  D);

    // encode x_start -> z (+ zbf mirror) ; x_end -> ze
    convert_f32_bf16<<<(B * Dx) / 1024, blk, 0, stream>>>(x_start, xbf, B * Dx);
    gemm_bf16<<<g1, blk, 0, stream>>>(xbf, ew1t, enc_b1, h, nullptr, B, Dx, H, 0);
    gemm_bf16<<<g2, blk, 0, stream>>>(h, ew2t, enc_b2, zbf, z, B, H, D, 1);
    convert_f32_bf16<<<(B * Dx) / 1024, blk, 0, stream>>>(x_end, xbf, B * Dx);
    gemm_bf16<<<g1, blk, 0, stream>>>(xbf, ew1t, enc_b1, h, nullptr, B, Dx, H, 0);
    gemm_bf16<<<g2, blk, 0, stream>>>(h, ew2t, enc_b2, nullptr, ze, B, H, D, 1);

    init_min_kernel<<<(B + 255) / 256, blk, 0, stream>>>(mind, B);

    for (int t = 0; t < STEPS; ++t) {
        gemm_bf16<<<g1, blk, 0, stream>>>(zbf, ww1t, wind_b1, h, nullptr, B, D, H, 0);
        gemm_bf16<<<g2, blk, 0, stream>>>(h, ww2t, wind_b2, zbf, z, B, H, D, 2);
        dist_min_kernel<<<B / 8, blk, 0, stream>>>(z, ze, mind);
    }

    mean_kernel<<<1, blk, 0, stream>>>(mind, (float*)d_out, B);
}
